// BackboneR3Denoiser_32933809226370
// MI455X (gfx1250) — matrix operand order repaired and checked by reading, not compiled, not run on hardware
//
#include <hip/hip_runtime.h>
#include <hip/hip_bf16.h>

#define B_   4
#define L_   1024
#define N_   4096
#define KNN_ 30
#define INVS_ 10
#define K_   40
#define E_   (N_ * K_)
#define NL_  4

typedef __attribute__((ext_vector_type(16))) _Float16 v16h;
typedef __attribute__((ext_vector_type(8)))  _Float16 h8;
typedef __attribute__((ext_vector_type(8)))  float    v8f;

// ----------------------------------------------------------------------------
// WMMA 16x16x32 f16->f32 fragment helpers (layouts per CDNA5 ISA 7.12.2).
// All buffers are K-padded so loads are unconditional, 16-byte vectorized.
// ----------------------------------------------------------------------------
__device__ __forceinline__ v16h load_a_frag(const _Float16* __restrict__ A, int lda,
                                            int row0, int k0) {
  const int lane = threadIdx.x & 31;
  // lane m = row0 + (lane&15); lanes 16..31 shift K by +8
  const _Float16* ap = A + (size_t)(row0 + (lane & 15)) * lda + k0 + ((lane >> 4) << 3);
  h8 lo = *(const h8*)ap;          // K = kb + 0..7
  h8 hi = *(const h8*)(ap + 16);   // K = kb + 16..23
  return __builtin_shufflevector(lo, hi, 0, 1, 2, 3, 4, 5, 6, 7,
                                         8, 9, 10, 11, 12, 13, 14, 15);
}

// B stored transposed: Bt[n][k], ldbT = padded K
__device__ __forceinline__ v16h load_b_frag(const _Float16* __restrict__ Bt, int ldbT,
                                            int k0, int n0) {
  const int lane = threadIdx.x & 31;
  const _Float16* bp = Bt + (size_t)(n0 + (lane & 15)) * ldbT + k0 + ((lane >> 4) << 3);
  h8 lo = *(const h8*)bp;          // K = kb + 0..7
  h8 hi = *(const h8*)(bp + 16);   // K = kb + 16..23
  return __builtin_shufflevector(lo, hi, 0, 1, 2, 3, 4, 5, 6, 7,
                                         8, 9, 10, 11, 12, 13, 14, 15);
}

// Generic GEMM: C[M x 32] = act(A[M x Kpad] @ B[Kpad x 32] + bias (+ C))
// blockDim = (32,4): one wave computes a full 16x32 strip (2 accumulators).
template<int RELU, int ACCUM, int OUTH>
__global__ void k_wmma_gemm(const _Float16* __restrict__ A, int lda,
                            const _Float16* __restrict__ Bt, int ldbT, int Kdim,
                            const float* __restrict__ bias,
                            float* __restrict__ Cf, _Float16* __restrict__ Ch,
                            int ldc, int Mrows) {
  const int mt   = blockIdx.x * blockDim.y + threadIdx.y;
  const int row0 = mt * 16;
  if (row0 >= Mrows) return;
  const int lane  = threadIdx.x & 31;
  const int ncl   = lane & 15;
  const int rbase = row0 + ((lane >> 4) << 3);

  v8f acc0 = {0.f, 0.f, 0.f, 0.f, 0.f, 0.f, 0.f, 0.f};
  v8f acc1 = {0.f, 0.f, 0.f, 0.f, 0.f, 0.f, 0.f, 0.f};
  if (ACCUM) {
#pragma unroll
    for (int r = 0; r < 8; ++r) {
      acc0[r] = Cf[(size_t)(rbase + r) * ldc + ncl];
      acc1[r] = Cf[(size_t)(rbase + r) * ldc + 16 + ncl];
    }
  }
  for (int k0 = 0; k0 < Kdim; k0 += 32) {
    if (k0 + 32 < Kdim)
      __builtin_prefetch(A + (size_t)(row0 + ncl) * lda + k0 + 32, 0, 0);
    v16h a  = load_a_frag(A, lda, row0, k0);
    v16h b0 = load_b_frag(Bt, ldbT, k0, 0);
    v16h b1 = load_b_frag(Bt, ldbT, k0, 16);
    acc0 = __builtin_amdgcn_wmma_f32_16x16x32_f16(false, a, false, b0,
                                                  (short)0, acc0, false, false);
    acc1 = __builtin_amdgcn_wmma_f32_16x16x32_f16(false, a, false, b1,
                                                  (short)0, acc1, false, false);
  }
  const float bv0 = bias ? bias[ncl]      : 0.0f;
  const float bv1 = bias ? bias[16 + ncl] : 0.0f;
#pragma unroll
  for (int r = 0; r < 8; ++r) {
    float v0 = acc0[r] + bv0;
    float v1 = acc1[r] + bv1;
    if (RELU) { v0 = fmaxf(v0, 0.0f); v1 = fmaxf(v1, 0.0f); }
    size_t off = (size_t)(rbase + r) * ldc;
    if (OUTH) { Ch[off + ncl] = (_Float16)v0; Ch[off + 16 + ncl] = (_Float16)v1; }
    else      { Cf[off + ncl] = v0;           Cf[off + 16 + ncl] = v1; }
  }
}

// ----------------------------------------------------------------------------
// Weight prep: transpose K x 32 f32 -> 32 x Kpad f16 (zero padded)
// ----------------------------------------------------------------------------
__global__ void k_wprep(const float* __restrict__ W, _Float16* __restrict__ Wt,
                        int K, int Kpad, int nmat) {
  int id = blockIdx.x * blockDim.x + threadIdx.x;
  if (id >= nmat * 32 * Kpad) return;
  int kp = id % Kpad;
  int rest = id / Kpad;
  int n = rest & 31;
  int mat = rest >> 5;
  float v = (kp < K) ? W[(size_t)mat * K * 32 + (size_t)kp * 32 + n] : 0.0f;
  Wt[(size_t)mat * 32 * Kpad + (size_t)n * Kpad + kp] = (_Float16)v;
}

__global__ void k_fill0(float* __restrict__ p, int n) {
  int i = blockIdx.x * blockDim.x + threadIdx.x;
  if (i < n) p[i] = 0.0f;
}

// Time embedding MLP: et[B][64]
__global__ void k_time(const float* __restrict__ t, const float* __restrict__ kappa,
                       const float* __restrict__ tW1, const float* __restrict__ tb1,
                       const float* __restrict__ tW2, const float* __restrict__ tb2,
                       float* __restrict__ et) {
  __shared__ float ft[64];
  __shared__ float h[128];
  const int tid = threadIdx.x;   // 128 threads
  for (int b = 0; b < B_; ++b) {
    if (tid < 64) {
      int c = tid & 31;
      float tp = 2.0f * 3.14159265358979f * t[b] * kappa[c];
      ft[tid] = (tid < 32) ? cosf(tp) : sinf(tp);
    }
    __syncthreads();
    {
      float s = tb1[tid];
      for (int c = 0; c < 64; ++c) s += ft[c] * tW1[c * 128 + tid];
      h[tid] = fmaxf(s, 0.0f);
    }
    __syncthreads();
    if (tid < 64) {
      float s = tb2[tid];
      for (int j = 0; j < 128; ++j) s += h[j] * tW2[j * 64 + tid];
      et[b * 64 + tid] = fmaxf(s, 0.0f);
    }
    __syncthreads();
  }
}

// Per-batch masked centroid of X0 = noised_bb[:,1]
__global__ void k_center(const float* __restrict__ nbb, const unsigned char* __restrict__ xmask,
                         float* __restrict__ center) {
  const int b = blockIdx.x, tid = threadIdx.x;
  __shared__ float sx[256], sy[256], sz[256], sw[256];
  float ax = 0, ay = 0, az = 0, aw = 0;
  for (int j = tid; j < L_; j += 256) {
    int n = b * L_ + j;
    float w = xmask[n] ? 0.0f : 1.0f;
    ax += w * nbb[n * 12 + 3];
    ay += w * nbb[n * 12 + 4];
    az += w * nbb[n * 12 + 5];
    aw += w;
  }
  sx[tid] = ax; sy[tid] = ay; sz[tid] = az; sw[tid] = aw;
  __syncthreads();
  for (int s = 128; s > 0; s >>= 1) {
    if (tid < s) { sx[tid] += sx[tid + s]; sy[tid] += sy[tid + s];
                   sz[tid] += sz[tid + s]; sw[tid] += sw[tid + s]; }
    __syncthreads();
  }
  if (tid == 0) {
    float w = fmaxf(sw[0], 1.0f);
    center[b * 3 + 0] = sx[0] / w;
    center[b * 3 + 1] = sy[0] / w;
    center[b * 3 + 2] = sz[0] / w;
  }
}

// X = X0 - center ; bb_rel from atoms {0,2,3}
__global__ void k_xinit(const float* __restrict__ nbb, const float* __restrict__ center,
                        float* __restrict__ X, float* __restrict__ bb_rel) {
  int n = blockIdx.x * blockDim.x + threadIdx.x;
  if (n >= N_) return;
  int b = n >> 10;
  for (int k = 0; k < 3; ++k) X[n * 3 + k] = nbb[n * 12 + 3 + k] - center[b * 3 + k];
  const int aoff[3] = {0, 6, 9};
  for (int a = 0; a < 3; ++a)
    for (int k = 0; k < 3; ++k)
      bb_rel[n * 9 + a * 3 + k] = nbb[n * 12 + aoff[a] + k];
}

// Build l0 GEMM A matrix: [feats0 | et(batch)]  (N x 96 f16)
__global__ void k_build_al0(const float* __restrict__ feats, const float* __restrict__ et,
                            _Float16* __restrict__ Al0) {
  int i = blockIdx.x * blockDim.x + threadIdx.x;
  if (i >= N_ * 96) return;
  int n = i / 96, k = i - n * 96;
  float v = (k < 32) ? feats[(size_t)n * 288 + k] : et[(n >> 10) * 64 + (k - 32)];
  Al0[i] = (_Float16)v;
}

// Pack nf rows (f16, K padded 35->64) + inv (f32, padded to 36)
__global__ void k_pack(const float* __restrict__ feats, const float* __restrict__ bb_rel,
                       const unsigned char* __restrict__ nmask,
                       _Float16* __restrict__ nfh, float* __restrict__ inv) {
  int t = blockIdx.x * blockDim.x + threadIdx.x;
  if (t >= N_ * 9) return;
  int n = t / 9, m = t - n * 9;
  float nm = nmask[n] ? 1.0f : 0.0f;
  _Float16* dst = nfh + (size_t)n * 576 + m * 64;
  for (int k = 0; k < 64; ++k) {
    float v = 0.0f;
    if (k < 32)                          v = feats[(size_t)n * 288 + m * 32 + k];
    else if (m == 0 && k == 34)          v = nm;
    else if (m >= 1 && m <= 3 && k < 35) v = bb_rel[n * 9 + (k - 32) * 3 + (m - 1)];
    dst[k] = (_Float16)v;
  }
  if (m == 0) {
    float* iv = inv + (size_t)n * 36;
    for (int k = 0; k < 36; ++k) {
      float v = 0.0f;
      if (k < 32) v = feats[(size_t)n * 288 + k];
      else if (k == 34) v = nm;
      iv[k] = v;
    }
  }
}

__device__ __forceinline__ float rng_u(unsigned a, unsigned b, unsigned c) {
  unsigned h = a * 0x9E3779B9u ^ (b + 0x7F4A7C15u) * 0x85EBCA6Bu ^
               (c + 0x165667B1u) * 0xC2B2AE35u;
  h ^= h >> 16; h *= 0x7FEB352Du; h ^= h >> 15; h *= 0x846CA68Bu; h ^= h >> 16;
  float u = (float)(h >> 8) * (1.0f / 16777216.0f);
  return fminf(fmaxf(u, 1e-6f), 1.0f - 1e-6f);
}

// KNN(30) + Gumbel-sampled(10) neighbor selection. One 256-thread block / node.
__global__ void k_edges(const float* __restrict__ X, const unsigned char* __restrict__ xmask,
                        int* __restrict__ nb, int layer) {
  const int n = blockIdx.x, b = n >> 10, base = b * L_;
  const int tid = threadIdx.x;
  __shared__ float sd[L_];
  __shared__ float rv[256];
  __shared__ int   ri[256];
  __shared__ int   sel[K_];
  float px = X[n * 3 + 0], py = X[n * 3 + 1], pz = X[n * 3 + 2];
  if (xmask[n]) { px = py = pz = 1e9f; }
  for (int j = tid; j < L_; j += 256) {
    int g = base + j;
    float qx = X[g * 3 + 0], qy = X[g * 3 + 1], qz = X[g * 3 + 2];
    if (xmask[g]) { qx = qy = qz = 1e9f; }
    float dx = qx - px, dy = qy - py, dz = qz - pz;
    sd[j] = sqrtf(dx * dx + dy * dy + dz * dz);
  }
  __syncthreads();
  // 30 arg-min extractions (stable tie-break on index)
  for (int it = 0; it < KNN_; ++it) {
    float bv = 3e38f; int bi = 1 << 30;
    for (int j = tid; j < L_; j += 256) {
      float v = sd[j];
      if (v < bv || (v == bv && j < bi)) { bv = v; bi = j; }
    }
    rv[tid] = bv; ri[tid] = bi; __syncthreads();
    for (int s = 128; s > 0; s >>= 1) {
      if (tid < s) {
        if (rv[tid + s] < rv[tid] || (rv[tid + s] == rv[tid] && ri[tid + s] < ri[tid])) {
          rv[tid] = rv[tid + s]; ri[tid] = ri[tid + s];
        }
      }
      __syncthreads();
    }
    if (tid == 0) { sel[it] = ri[0]; sd[ri[0]] = 3e38f; }
    __syncthreads();
  }
  // Gumbel keys for remaining candidates
  for (int j = tid; j < L_; j += 256) {
    float v = sd[j];
    if (v < 1e38f) {
      float u = rng_u((unsigned)layer, (unsigned)n, (unsigned)j);
      sd[j] = -3.0f * logf(fmaxf(v, 1e-9f)) - logf(-logf(u));
    } else sd[j] = -3e38f;
  }
  __syncthreads();
  for (int it = 0; it < INVS_; ++it) {
    float bv = -3e38f; int bi = 1 << 30;
    for (int j = tid; j < L_; j += 256) {
      float v = sd[j];
      if (v > bv || (v == bv && j < bi)) { bv = v; bi = j; }
    }
    rv[tid] = bv; ri[tid] = bi; __syncthreads();
    for (int s = 128; s > 0; s >>= 1) {
      if (tid < s) {
        if (rv[tid + s] > rv[tid] || (rv[tid + s] == rv[tid] && ri[tid + s] < ri[tid])) {
          rv[tid] = rv[tid + s]; ri[tid] = ri[tid + s];
        }
      }
      __syncthreads();
    }
    if (tid == 0) { sel[KNN_ + it] = ri[0]; sd[ri[0]] = -3e38f; }
    __syncthreads();
  }
  if (tid < K_) nb[n * K_ + tid] = base + sel[tid];
}

// Per-edge RBF + positional embedding (E x 32 f16) + validity flag
__global__ void k_phi(const float* __restrict__ X, const int* __restrict__ nb,
                      _Float16* __restrict__ phi, unsigned char* __restrict__ valid) {
  int e = blockIdx.x * blockDim.x + threadIdx.x;
  if (e >= E_) return;
  int n = e / K_;
  int g = nb[e];
  float dx = X[g * 3 + 0] - X[n * 3 + 0];
  float dy = X[g * 3 + 1] - X[n * 3 + 1];
  float dz = X[g * 3 + 2] - X[n * 3 + 2];
  float dist = sqrtf(dx * dx + dy * dy + dz * dz);
  valid[e] = (dist > 0.1f && dist < 1e8f) ? 1 : 0;
  _Float16* p = phi + (size_t)e * 32;
  for (int gg = 0; gg < 16; ++gg) {
    float mu = (float)gg * (20.0f / 15.0f);
    float z = (dist - mu) / 1.25f;
    p[gg] = (_Float16)expf(-z * z);
  }
  float fd = (float)(g - n);
  for (int f = 0; f < 8; ++f) {
    float fr = expf(-(float)(2 * f) * (logf(10000.0f) / 16.0f));
    float a = fd * fr;
    p[16 + f] = (_Float16)cosf(a);
    p[24 + f] = (_Float16)sinf(a);
  }
}

// Attention logits: per (edge, head) dot over 102 features
__global__ void k_logits(const float* __restrict__ inv, const _Float16* __restrict__ ebuf,
                         const int* __restrict__ nb, const unsigned char* __restrict__ valid,
                         const float* __restrict__ Wa, const float* __restrict__ ba,
                         float* __restrict__ logits) {
  int id = blockIdx.x * blockDim.x + threadIdx.x;
  if (id >= E_ * 8) return;
  int e = id >> 3, head = id & 7;
  int n = e / K_, g = nb[e];
  float s = ba[head];
  const float* ivn = inv + (size_t)g * 36;
  const float* ivs = inv + (size_t)n * 36;
  for (int k = 0; k < 35; ++k) s += ivn[k] * Wa[k * 8 + head];
  for (int k = 0; k < 35; ++k) s += ivs[k] * Wa[(35 + k) * 8 + head];
  const _Float16* ee = ebuf + (size_t)e * 32;
  for (int k = 0; k < 32; ++k) s += (float)ee[k] * Wa[(70 + k) * 8 + head];
  logits[(size_t)e * 8 + head] = valid[e] ? s : -1e9f;
}

// Segment softmax over each node's contiguous 40-edge group, per head
__global__ void k_softmax(const float* __restrict__ logits, float* __restrict__ alpha) {
  int id = blockIdx.x * blockDim.x + threadIdx.x;
  if (id >= N_ * 8) return;
  int n = id >> 3, head = id & 7;
  size_t base = (size_t)n * K_;
  float mx = -3e38f;
  for (int k = 0; k < K_; ++k) mx = fmaxf(mx, logits[(base + k) * 8 + head]);
  float sum = 0.0f;
  for (int k = 0; k < K_; ++k) sum += expf(logits[(base + k) * 8 + head] - mx);
  float inv_s = 1.0f / (sum + 1e-9f);
  for (int k = 0; k < K_; ++k)
    alpha[(base + k) * 8 + head] = expf(logits[(base + k) * 8 + head] - mx) * inv_s;
}

// agg[n,m,c] = sum_k v[nb[n,k], m, c] * alpha[n,k, c/4]   -> f16 for next GEMM
__global__ void k_agg(const float* __restrict__ vbuf, const float* __restrict__ alpha,
                      const int* __restrict__ nb, _Float16* __restrict__ aggh) {
  int id = blockIdx.x * blockDim.x + threadIdx.x;
  if (id >= N_ * 288) return;
  int n = id / 288, mc = id - n * 288;
  int head = (mc & 31) >> 2;
  float s = 0.0f;
  for (int k = 0; k < K_; ++k) {
    int e = n * K_ + k;
    s += vbuf[(size_t)nb[e] * 288 + mc] * alpha[(size_t)e * 8 + head];
  }
  aggh[id] = (_Float16)s;
}

// Extract feats row0 as f16 for the FFN GEMMs
__global__ void k_row0h(const float* __restrict__ feats, _Float16* __restrict__ f0h) {
  int i = blockIdx.x * blockDim.x + threadIdx.x;
  if (i >= N_ * 32) return;
  f0h[i] = (_Float16)feats[(size_t)(i >> 5) * 288 + (i & 31)];
}

// Per-node geometry update: upd/gate/ub -> X, bb_rel
__global__ void k_epilogue(const float* __restrict__ feats,
                           const float* __restrict__ Wx1, const float* __restrict__ Wg,
                           const float* __restrict__ bg,  const float* __restrict__ Wb1,
                           const unsigned char* __restrict__ nmask,
                           float* __restrict__ X, float* __restrict__ bb_rel) {
  int n = blockIdx.x * blockDim.x + threadIdx.x;
  if (n >= N_) return;
  const float* o = feats + (size_t)n * 288;
  float gi = bg[0];
  for (int c = 0; c < 32; ++c) gi += o[c] * Wg[c];
  float gate = (gi > 20.0f) ? gi : log1pf(expf(gi));
  bool nm = nmask[n] != 0;
  for (int j = 0; j < 3; ++j) {
    const float* orow = o + (1 + j) * 32;
    float upd = 0.0f;
    for (int c = 0; c < 32; ++c) upd += orow[c] * Wx1[c];
    if (nm) X[n * 3 + j] += upd * gate;
    for (int a = 0; a < 3; ++a) {
      float ub = 0.0f;
      for (int c = 0; c < 32; ++c) ub += orow[c] * Wb1[c * 3 + a];
      if (nm) bb_rel[n * 9 + a * 3 + j] += ub;
    }
  }
}

__global__ void k_final(const float* __restrict__ X, const float* __restrict__ center,
                        const float* __restrict__ bb_rel, float* __restrict__ den) {
  int n = blockIdx.x * blockDim.x + threadIdx.x;
  if (n >= N_) return;
  int b = n >> 10;
  const int aoff[3] = {0, 6, 9};
  for (int k = 0; k < 3; ++k)
    den[n * 12 + 3 + k] = X[n * 3 + k] + center[b * 3 + k];
  for (int a = 0; a < 3; ++a)
    for (int k = 0; k < 3; ++k)
      den[n * 12 + aoff[a] + k] = bb_rel[n * 9 + a * 3 + k];
}

// ----------------------------------------------------------------------------
extern "C" void kernel_launch(void* const* d_in, const int* in_sizes, int n_in,
                              void* d_out, int out_size, void* d_ws, size_t ws_size,
                              hipStream_t stream) {
  (void)in_sizes; (void)n_in; (void)out_size; (void)ws_size;
  const float* nbb   = (const float*)d_in[0];
  const float* t     = (const float*)d_in[1];
  const unsigned char* xmask = (const unsigned char*)d_in[2];
  const unsigned char* nmask = (const unsigned char*)d_in[3];
  const float* kappa = (const float*)d_in[4];
  const float* tW1 = (const float*)d_in[5];
  const float* tb1 = (const float*)d_in[6];
  const float* tW2 = (const float*)d_in[7];
  const float* tb2 = (const float*)d_in[8];
  const float* eW  = (const float*)d_in[9];
  const float* eb  = (const float*)d_in[10];
  const float* We  = (const float*)d_in[11];
  const float* be  = (const float*)d_in[12];
  const float* Wa  = (const float*)d_in[13];
  const float* ba  = (const float*)d_in[14];
  const float* Wv  = (const float*)d_in[15];
  const float* bv  = (const float*)d_in[16];
  const float* Wo  = (const float*)d_in[17];
  const float* bo  = (const float*)d_in[18];
  const float* Wf1 = (const float*)d_in[19];
  const float* bf1 = (const float*)d_in[20];
  const float* Wf2 = (const float*)d_in[21];
  const float* bf2 = (const float*)d_in[22];
  const float* Wx  = (const float*)d_in[23];
  const float* Wg  = (const float*)d_in[25];
  const float* bg  = (const float*)d_in[26];
  const float* Wb  = (const float*)d_in[27];
  float* den = (float*)d_out;

  // ---- workspace bump allocator (256-byte aligned blocks) ----
  char* base = (char*)d_ws;
  size_t off = 0;
  auto alloc = [&](size_t bytes) -> char* {
    char* p = base + off;
    off = (off + bytes + 255) & ~(size_t)255;
    return p;
  };
  float*     X      = (float*)alloc(N_ * 3 * sizeof(float));
  float*     center = (float*)alloc(B_ * 3 * sizeof(float));
  float*     et     = (float*)alloc(B_ * 64 * sizeof(float));
  float*     feats  = (float*)alloc((size_t)N_ * 288 * sizeof(float));
  float*     bb_rel = (float*)alloc(N_ * 9 * sizeof(float));
  float*     inv    = (float*)alloc((size_t)N_ * 36 * sizeof(float));
  _Float16*  nfh    = (_Float16*)alloc((size_t)N_ * 576 * sizeof(_Float16));
  _Float16*  Al0    = (_Float16*)alloc((size_t)N_ * 96 * sizeof(_Float16));
  int*       nb     = (int*)alloc((size_t)E_ * sizeof(int));
  unsigned char* valid = (unsigned char*)alloc(E_);
  _Float16*  phi    = (_Float16*)alloc((size_t)E_ * 32 * sizeof(_Float16));
  _Float16*  ebuf   = (_Float16*)alloc((size_t)E_ * 32 * sizeof(_Float16));
  float*     logits = (float*)alloc((size_t)E_ * 8 * sizeof(float));
  float*     alpha  = (float*)alloc((size_t)E_ * 8 * sizeof(float));
  float*     vbuf   = (float*)alloc((size_t)N_ * 288 * sizeof(float));
  _Float16*  aggh   = (_Float16*)alloc((size_t)N_ * 288 * sizeof(_Float16));
  _Float16*  f0h    = (_Float16*)alloc((size_t)N_ * 32 * sizeof(_Float16));
  _Float16*  h1h    = (_Float16*)alloc((size_t)N_ * 32 * sizeof(_Float16));
  // transposed + K-padded f16 weights
  _Float16*  eWt    = (_Float16*)alloc(32 * 96 * sizeof(_Float16));            // K=96
  _Float16*  Wet    = (_Float16*)alloc((size_t)NL_ * 32 * 32 * sizeof(_Float16));
  _Float16*  Wvt    = (_Float16*)alloc((size_t)NL_ * 3 * 32 * 64 * sizeof(_Float16)); // K 35->64
  _Float16*  Wot    = (_Float16*)alloc((size_t)NL_ * 3 * 32 * 32 * sizeof(_Float16));
  _Float16*  Wf1t   = (_Float16*)alloc((size_t)NL_ * 32 * 32 * sizeof(_Float16));
  _Float16*  Wf2t   = (_Float16*)alloc((size_t)NL_ * 32 * 32 * sizeof(_Float16));

  auto cdiv = [](int a, int b) { return (a + b - 1) / b; };

  // ---- weight prep: transpose + pad + f16 ----
  k_wprep<<<cdiv(32 * 96, 256), 256, 0, stream>>>(eW, eWt, 96, 96, 1);
  k_wprep<<<cdiv(NL_ * 1024, 256), 256, 0, stream>>>(We, Wet, 32, 32, NL_);
  k_wprep<<<cdiv(NL_ * 3 * 2048, 256), 256, 0, stream>>>(Wv, Wvt, 35, 64, NL_ * 3);
  k_wprep<<<cdiv(NL_ * 3 * 1024, 256), 256, 0, stream>>>(Wo, Wot, 32, 32, NL_ * 3);
  k_wprep<<<cdiv(NL_ * 1024, 256), 256, 0, stream>>>(Wf1, Wf1t, 32, 32, NL_);
  k_wprep<<<cdiv(NL_ * 1024, 256), 256, 0, stream>>>(Wf2, Wf2t, 32, 32, NL_);

  // ---- setup ----
  k_time<<<1, 128, 0, stream>>>(t, kappa, tW1, tb1, tW2, tb2, et);
  k_center<<<B_, 256, 0, stream>>>(nbb, xmask, center);
  k_xinit<<<cdiv(N_, 256), 256, 0, stream>>>(nbb, center, X, bb_rel);
  k_fill0<<<cdiv(N_ * 288, 256), 256, 0, stream>>>(feats, N_ * 288);

  const dim3 gblock(32, 4, 1);
  const int  lmap[9] = {0, 1, 1, 1, 2, 2, 2, 2, 2};
  const dim3 gridN(cdiv(N_ / 16, 4), 1, 1);   // 4096 rows x 32 cols
  const dim3 gridE(cdiv(E_ / 16, 4), 1, 1);   // 163840 rows x 32 cols

  for (int i = 0; i < NL_; ++i) {
    // node linear: feats0 = [feats0|et] @ eW + eb
    k_build_al0<<<cdiv(N_ * 96, 256), 256, 0, stream>>>(feats, et, Al0);
    k_wmma_gemm<0, 0, 0><<<gridN, gblock, 0, stream>>>(
        Al0, 96, eWt, 96, 96, eb, feats, (_Float16*)nullptr, 288, N_);

    // pack nf / inv
    k_pack<<<cdiv(N_ * 9, 256), 256, 0, stream>>>(feats, bb_rel, nmask, nfh, inv);

    // neighbor sampling
    k_edges<<<N_, 256, 0, stream>>>(X, xmask, nb, i);

    // edge features: phi -> e = relu(phi @ We + be)
    k_phi<<<cdiv(E_, 256), 256, 0, stream>>>(X, nb, phi, valid);
    k_wmma_gemm<1, 0, 1><<<gridE, gblock, 0, stream>>>(
        phi, 32, Wet + i * 1024, 32, 32, be + i * 32,
        (float*)nullptr, ebuf, 32, E_);

    // attention
    k_logits<<<cdiv(E_ * 8, 256), 256, 0, stream>>>(
        inv, ebuf, nb, valid, Wa + i * 102 * 8, ba + i * 8, logits);
    k_softmax<<<cdiv(N_ * 8, 256), 256, 0, stream>>>(logits, alpha);

    // value projection: 9 SO(3) GEMMs (K padded 35->64)
    for (int m = 0; m < 9; ++m) {
      k_wmma_gemm<0, 0, 0><<<gridN, gblock, 0, stream>>>(
          nfh + m * 64, 576, Wvt + (i * 3 + lmap[m]) * 2048, 64, 64,
          (m == 0) ? (bv + i * 32) : nullptr,
          vbuf + m * 32, (_Float16*)nullptr, 288, N_);
    }

    // message aggregation
    k_agg<<<cdiv(N_ * 288, 256), 256, 0, stream>>>(vbuf, alpha, nb, aggh);

    // output projection: 9 SO(3) GEMMs, written into feats (feats = out)
    for (int m = 0; m < 9; ++m) {
      k_wmma_gemm<0, 0, 0><<<gridN, gblock, 0, stream>>>(
          aggh + m * 32, 288, Wot + (i * 3 + lmap[m]) * 1024, 32, 32,
          (m == 0) ? (bo + i * 32) : nullptr,
          feats + m * 32, (_Float16*)nullptr, 288, N_);
    }

    // FFN on row 0 (residual accumulate into feats row 0)
    k_row0h<<<cdiv(N_ * 32, 256), 256, 0, stream>>>(feats, f0h);
    k_wmma_gemm<1, 0, 1><<<gridN, gblock, 0, stream>>>(
        f0h, 32, Wf1t + i * 1024, 32, 32, bf1 + i * 32,
        (float*)nullptr, h1h, 32, N_);
    k_wmma_gemm<0, 1, 0><<<gridN, gblock, 0, stream>>>(
        h1h, 32, Wf2t + i * 1024, 32, 32, bf2 + i * 32,
        feats, (_Float16*)nullptr, 288, N_);

    // geometry update
    k_epilogue<<<cdiv(N_, 256), 256, 0, stream>>>(
        feats, Wx + (i * 3 + 1) * 32, Wg + i * 32, bg + i,
        Wb + (i * 3 + 1) * 96, nmask, X, bb_rel);
  }

  k_final<<<cdiv(N_, 256), 256, 0, stream>>>(X, center, bb_rel, den);
}
